// MSDeformAttnGrid_convoffsets_30124900614204
// MI455X (gfx1250) — matrix operand order repaired and checked by reading, not compiled, not run on hardware
//
#include <hip/hip_runtime.h>
#include <hip/hip_bf16.h>
#include <math.h>

// Problem constants (from reference)
#define Nn   2
#define Cc   256
#define Hh   128
#define Wcol 128
#define NHh  8
#define Pp   4
#define DHh  32
#define HPp  130
#define WPp  130
#define LQq  16384          // H*W
#define Mrows (Nn*LQq)      // 32768
#define Kk   256
#define KPAD (Kk + 8)       // LDS row stride (elements): 528B -> bank offset 4*row, conflict-free

typedef __attribute__((ext_vector_type(16))) __bf16 bf16x16;
typedef __attribute__((ext_vector_type(8)))  float  f32x8;

__device__ __forceinline__ unsigned short f2bf(float f) {
  unsigned int u = __float_as_uint(f);
  u += 0x7FFFu + ((u >> 16) & 1u);           // round-to-nearest-even
  return (unsigned short)(u >> 16);
}

// ---------------------------------------------------------------------------
// Kernel 1: depthwise 7x7 conv + channel LayerNorm + exact GELU -> bf16 acts
// one block per pixel (n,h,w); thread = channel (coalesced on C)
// ---------------------------------------------------------------------------
__global__ void __launch_bounds__(256)
dwconv_ln_gelu(const float* __restrict__ query,
               const float* __restrict__ dwk,
               const float* __restrict__ dwb,
               const float* __restrict__ lnw,
               const float* __restrict__ lnb,
               unsigned short* __restrict__ q_bf) {
  const int pix = blockIdx.x;                // n*LQ + l
  const int n = pix >> 14;
  const int l = pix & (LQq - 1);
  const int h = l >> 7;
  const int w = l & (Wcol - 1);
  const int c = threadIdx.x;

  float kern[49];
  #pragma unroll
  for (int t = 0; t < 49; ++t) kern[t] = dwk[c * 49 + t];

  float acc = dwb[c];
  #pragma unroll
  for (int ky = 0; ky < 7; ++ky) {
    const int hh = h + ky - 3;
    if (hh < 0 || hh >= Hh) continue;
    #pragma unroll
    for (int kx = 0; kx < 7; ++kx) {
      const int ww = w + kx - 3;
      if (ww < 0 || ww >= Wcol) continue;
      acc += query[((size_t)n * LQq + hh * Wcol + ww) * Cc + c] * kern[ky * 7 + kx];
    }
  }

  __shared__ float s1[256];
  __shared__ float s2[256];
  s1[c] = acc;
  s2[c] = acc * acc;
  __syncthreads();
  for (int off = 128; off > 0; off >>= 1) {
    if (c < off) { s1[c] += s1[c + off]; s2[c] += s2[c + off]; }
    __syncthreads();
  }
  const float mu  = s1[0] * (1.0f / 256.0f);
  const float var = s2[0] * (1.0f / 256.0f) - mu * mu;

  float x = (acc - mu) * rsqrtf(var + 1e-5f) * lnw[c] + lnb[c];
  x = 0.5f * x * (1.0f + erff(x * 0.70710678118654752f));   // exact GELU
  q_bf[(size_t)pix * Cc + c] = f2bf(x);
}

// ---------------------------------------------------------------------------
// Kernel 2: weight fp32 -> bf16 with transpose: dst[nc*K + k] = src[k*Nc + nc]
// ---------------------------------------------------------------------------
__global__ void transpose_w_bf(const float* __restrict__ src,
                               unsigned short* __restrict__ dst, int Ncols) {
  const int i = blockIdx.x * blockDim.x + threadIdx.x;
  if (i >= Ncols * Kk) return;
  const int nc = i / Kk;
  const int k  = i % Kk;
  dst[i] = f2bf(src[k * Ncols + nc]);
}

// ---------------------------------------------------------------------------
// Kernel 3: zero-fill (for padded value tensor borders)
// ---------------------------------------------------------------------------
__global__ void zero_f4(float4* __restrict__ p, int n4) {
  const int i = blockIdx.x * blockDim.x + threadIdx.x;
  if (i < n4) p[i] = make_float4(0.f, 0.f, 0.f, 0.f);
}

// ---------------------------------------------------------------------------
// Epilogue helper shared by both GEMM kernels.
// mode: 0 = +bias plain, 1 = sigmoid(+bias), 2 = scatter into padded value
// ---------------------------------------------------------------------------
__device__ __forceinline__ void gemm_epilogue(const f32x8& c0, const f32x8& c1,
                                              const float* __restrict__ bias,
                                              float* __restrict__ out,
                                              int row0, int col0, int lane,
                                              int Ncols, int mode) {
  const int rbase = (lane >> 4) << 3;
  const int cA    = lane & 15;
  #pragma unroll
  for (int r = 0; r < 8; ++r) {
    const int row = row0 + rbase + r;
    const int cL = col0 + cA;
    const int cH = cL + 16;
    float v0 = c0[r] + bias[cL];
    float v1 = c1[r] + bias[cH];
    if (mode == 1) {
      v0 = 1.0f / (1.0f + __expf(-v0));
      v1 = 1.0f / (1.0f + __expf(-v1));
    }
    if (mode == 2) {
      const int n = row >> 14;
      const int l = row & (LQq - 1);
      const int h = l >> 7;
      const int w = l & (Wcol - 1);
      const size_t base = ((size_t)(n * HPp + h + 1) * WPp + (w + 1)) * Cc;
      out[base + cL] = v0;
      out[base + cH] = v1;
    } else {
      out[(size_t)row * Ncols + cL] = v0;
      out[(size_t)row * Ncols + cH] = v1;
    }
  }
}

// ---------------------------------------------------------------------------
// Kernel 4a: WMMA GEMM with async global->LDS staging of the A tile.
// Requires Ncols % 64 == 0. Block = 8 waves = 4 M-tiles x 2 N-tiles (64x64).
// A tile (64 x 256 bf16) staged via GLOBAL_LOAD_ASYNC_TO_LDS_B128 (ASYNCcnt),
// then read back as WMMA fragments with ds_load_b128 (padded, conflict-free).
// ---------------------------------------------------------------------------
__global__ void __launch_bounds__(256)
gemm_bf16_wmma_staged(const unsigned short* __restrict__ A,
                      const unsigned short* __restrict__ Bt,
                      const float* __restrict__ bias,
                      float* __restrict__ out,
                      int Ncols, int mode) {
  __shared__ unsigned short As[64 * KPAD];   // 64 rows x 528B = 33 KB

  const int tid  = threadIdx.x;
  const int lane = tid & 31;
  const int wid  = tid >> 5;
  const int nbc  = Ncols >> 6;               // 64-wide column blocks
  const int br   = blockIdx.x / nbc;         // 64-row block
  const int bc   = blockIdx.x % nbc;

  // ---- async stage A tile: 64 rows x 256 bf16 = 2048 x b128 chunks ----
  #pragma unroll
  for (int it = 0; it < 8; ++it) {
    const int i   = it * 256 + tid;          // 0..2047
    const int row = i >> 5;
    const int ch  = i & 31;
    const unsigned short* gp = A + ((size_t)(br * 64 + row) * Kk + ch * 8);
    unsigned lds_off = (unsigned)(size_t)(&As[row * KPAD + ch * 8]);
    asm volatile("global_load_async_to_lds_b128 %0, %1, off"
                 :: "v"(lds_off), "v"(gp)
                 : "memory");
  }
  asm volatile("s_wait_asynccnt 0x0" ::: "memory");
  __syncthreads();

  // ---- per-wave 16x32 tile ----
  const int mt   = wid >> 1;                 // 0..3
  const int ntw  = wid & 1;                  // 0..1
  const int row0 = br * 64 + mt * 16;
  const int col0 = bc * 64 + ntw * 32;

  // A fragment: lane L -> local row mt*16 + (L&15), K = (L>>4)*8 + {0..7,16..23}
  const int lrow  = mt * 16 + (lane & 15);
  const int koffA = (lane >> 4) << 3;
  const unsigned short* aLds = &As[lrow * KPAD + koffA];

  // B fragment: lane L -> N = col0 + (L&15), K = (L>>4)*8 + {0..7,16..23}
  const int nB  = col0 + (lane & 15);
  const int kbB = (lane >> 4) << 3;
  const unsigned short* bptr0 = Bt + (size_t)nB * Kk + kbB;
  const unsigned short* bptr1 = Bt + (size_t)(nB + 16) * Kk + kbB;

  f32x8 c0 = {};
  f32x8 c1 = {};
  union U { uint4 u4[2]; bf16x16 v; };

  #pragma unroll
  for (int kt = 0; kt < Kk; kt += 32) {
    U a, b0, b1;
    a.u4[0]  = *(const uint4*)(aLds + kt);        // ds_load_b128
    a.u4[1]  = *(const uint4*)(aLds + kt + 16);
    b0.u4[0] = *(const uint4*)(bptr0 + kt);
    b0.u4[1] = *(const uint4*)(bptr0 + kt + 16);
    b1.u4[0] = *(const uint4*)(bptr1 + kt);
    b1.u4[1] = *(const uint4*)(bptr1 + kt + 16);
    c0 = __builtin_amdgcn_wmma_f32_16x16x32_bf16(false, a.v, false, b0.v,
                                                 (short)0, c0, false, false);
    c1 = __builtin_amdgcn_wmma_f32_16x16x32_bf16(false, a.v, false, b1.v,
                                                 (short)0, c1, false, false);
  }

  gemm_epilogue(c0, c1, bias, out, row0, col0, lane, Ncols, mode);
}

// ---------------------------------------------------------------------------
// Kernel 4b: direct-from-global WMMA GEMM (used for the narrow attn GEMM).
// ---------------------------------------------------------------------------
__global__ void __launch_bounds__(256)
gemm_bf16_wmma(const unsigned short* __restrict__ A,
               const unsigned short* __restrict__ Bt,
               const float* __restrict__ bias,
               float* __restrict__ out,
               int Ncols, int mode) {
  const int lane = threadIdx.x & 31;
  const int wid  = threadIdx.x >> 5;
  const int NtW  = Ncols >> 5;
  const int MtW  = Mrows >> 4;
  const int gw   = blockIdx.x * 8 + wid;
  if (gw >= MtW * NtW) return;               // wave-uniform exit
  const int tm = gw / NtW;
  const int tn = gw % NtW;
  const int row0 = tm << 4;
  const int col0 = tn << 5;

  const int mA    = lane & 15;
  const int koffA = (lane >> 4) << 3;
  const int nB  = col0 + (lane & 15);
  const int kbB = (lane >> 4) << 3;

  const unsigned short* aptr  = A  + (size_t)(row0 + mA) * Kk + koffA;
  const unsigned short* bptr0 = Bt + (size_t)nB * Kk + kbB;
  const unsigned short* bptr1 = Bt + (size_t)(nB + 16) * Kk + kbB;

  f32x8 c0 = {};
  f32x8 c1 = {};
  union U { uint4 u4[2]; bf16x16 v; };

  #pragma unroll
  for (int kt = 0; kt < Kk; kt += 32) {
    U a, b0, b1;
    a.u4[0]  = *(const uint4*)(aptr + kt);
    a.u4[1]  = *(const uint4*)(aptr + kt + 16);
    b0.u4[0] = *(const uint4*)(bptr0 + kt);
    b0.u4[1] = *(const uint4*)(bptr0 + kt + 16);
    b1.u4[0] = *(const uint4*)(bptr1 + kt);
    b1.u4[1] = *(const uint4*)(bptr1 + kt + 16);
    if (kt + 32 < Kk) {
      __builtin_prefetch(aptr + kt + 32, 0, 0);   // global_prefetch_b8
    }
    c0 = __builtin_amdgcn_wmma_f32_16x16x32_bf16(false, a.v, false, b0.v,
                                                 (short)0, c0, false, false);
    c1 = __builtin_amdgcn_wmma_f32_16x16x32_bf16(false, a.v, false, b1.v,
                                                 (short)0, c1, false, false);
  }

  gemm_epilogue(c0, c1, bias, out, row0, col0, lane, Ncols, mode);
}

// ---------------------------------------------------------------------------
// Kernel 5: deformable bilinear sampling + attention-weighted sum
// block = pixel (n,l); thread t -> head = t>>5, d = t&31 (channel = t)
// ---------------------------------------------------------------------------
__global__ void __launch_bounds__(256)
deform_sample(const float* __restrict__ value_pad,
              const float* __restrict__ offb,
              const float* __restrict__ attnb,
              const float* __restrict__ refp,
              unsigned short* __restrict__ samp_bf) {
  const int pix = blockIdx.x;
  const int n = pix >> 14;
  const int t = threadIdx.x;
  const int head = t >> 5;
  const int d = t & 31;

  const float refx = refp[(size_t)pix * 2 + 0];
  const float refy = refp[(size_t)pix * 2 + 1];
  const float* vp = value_pad + (size_t)n * (HPp * WPp * Cc) + head * DHh + d;

  float acc = 0.f;
  #pragma unroll
  for (int p = 0; p < Pp; ++p) {
    const float o0 = offb[(size_t)pix * 64 + head * 8 + p * 2 + 0];
    const float o1 = offb[(size_t)pix * 64 + head * 8 + p * 2 + 1];
    const float aw = attnb[(size_t)pix * 32 + head * 4 + p];
    const float x = refx * (float)WPp + (float)(p >> 1) + o0 - 0.5f;
    const float y = refy * (float)HPp + (float)(p & 1) + o1 - 0.5f;
    const float x0f = floorf(x), y0f = floorf(y);
    const float wx1 = x - x0f, wy1 = y - y0f;
    const float wx0 = 1.f - wx1, wy0 = 1.f - wy1;
    const int x0 = (int)x0f, y0 = (int)y0f;
    #pragma unroll
    for (int cy = 0; cy < 2; ++cy) {
      const int yi = y0 + cy;
      const float wy = cy ? wy1 : wy0;
      #pragma unroll
      for (int cx = 0; cx < 2; ++cx) {
        const int xi = x0 + cx;
        const float wx = cx ? wx1 : wx0;
        const bool valid = (xi >= 0) && (xi < WPp) && (yi >= 0) && (yi < HPp);
        const int xc = min(max(xi, 0), WPp - 1);
        const int yc = min(max(yi, 0), HPp - 1);
        const float g = valid ? vp[(size_t)(yc * WPp + xc) * Cc] : 0.f;
        acc += aw * wy * wx * g;
      }
    }
  }
  samp_bf[(size_t)pix * Cc + t] = f2bf(acc);
}

// ---------------------------------------------------------------------------
extern "C" void kernel_launch(void* const* d_in, const int* in_sizes, int n_in,
                              void* d_out, int out_size, void* d_ws, size_t ws_size,
                              hipStream_t stream) {
  (void)in_sizes; (void)n_in; (void)out_size; (void)ws_size;

  const float* query  = (const float*)d_in[0];
  const float* refp   = (const float*)d_in[1];
  // d_in[2]=spatial_shapes, d_in[3]=level_start_index: constants, unused
  const float* dwk    = (const float*)d_in[4];
  const float* dwb    = (const float*)d_in[5];
  const float* lnw    = (const float*)d_in[6];
  const float* lnb    = (const float*)d_in[7];
  const float* W_off  = (const float*)d_in[8];
  const float* b_off  = (const float*)d_in[9];
  const float* W_attn = (const float*)d_in[10];
  const float* b_attn = (const float*)d_in[11];
  const float* W_val  = (const float*)d_in[12];
  const float* b_val  = (const float*)d_in[13];
  const float* W_out  = (const float*)d_in[14];
  const float* b_out  = (const float*)d_in[15];
  float* out = (float*)d_out;

  // Workspace carve-up (all offsets 256B-aligned)
  char* ws = (char*)d_ws;
  unsigned short* q_bf    = (unsigned short*)(ws);                       // 16,777,216 B
  unsigned short* Wval_t  = (unsigned short*)(ws + 16777216);            //    131,072 B
  unsigned short* Woff_t  = (unsigned short*)(ws + 16908288);            //     32,768 B
  unsigned short* Wattn_t = (unsigned short*)(ws + 16941056);            //     16,384 B
  unsigned short* Wout_t  = (unsigned short*)(ws + 16957440);            //    131,072 B
  float*          val_pad = (float*)        (ws + 17088512);            // 34,611,200 B
  float*          offb    = (float*)        (ws + 51699712);            //  8,388,608 B
  float*          attnb   = (float*)        (ws + 60088320);            //  4,194,304 B
  unsigned short* samp_bf = (unsigned short*)(ws + 64282624);            // 16,777,216 B

  // 1) weight prep (bf16, transposed to K-contiguous)
  transpose_w_bf<<<(256 * Kk + 255) / 256, 256, 0, stream>>>(W_val,  Wval_t,  256);
  transpose_w_bf<<<( 64 * Kk + 255) / 256, 256, 0, stream>>>(W_off,  Woff_t,   64);
  transpose_w_bf<<<( 32 * Kk + 255) / 256, 256, 0, stream>>>(W_attn, Wattn_t,  32);
  transpose_w_bf<<<(256 * Kk + 255) / 256, 256, 0, stream>>>(W_out,  Wout_t,  256);

  // 2) dwconv + LN + GELU -> bf16 activations
  dwconv_ln_gelu<<<Mrows, 256, 0, stream>>>(query, dwk, dwb, lnw, lnb, q_bf);

  // 3) zero padded value tensor (borders must be 0)
  {
    const int n4 = (Nn * HPp * WPp * Cc) / 4;       // 2,163,200 float4
    zero_f4<<<(n4 + 255) / 256, 256, 0, stream>>>((float4*)val_pad, n4);
  }

  // 4) GEMMs from q: value (scatter into pad), offsets, attn (sigmoid)
  {
    const int nrb = Mrows / 64;                     // 512 row blocks
    gemm_bf16_wmma_staged<<<nrb * (256 / 64), 256, 0, stream>>>(q_bf, Wval_t, b_val, val_pad, 256, 2);
    gemm_bf16_wmma_staged<<<nrb * ( 64 / 64), 256, 0, stream>>>(q_bf, Woff_t, b_off, offb,     64, 0);
    // attn: only 32 columns -> direct path (Ncols % 64 != 0)
    gemm_bf16_wmma<<<((Mrows >> 4) * (32 >> 5)) / 8, 256, 0, stream>>>(q_bf, Wattn_t, b_attn, attnb, 32, 1);
  }

  // 5) deformable sampling -> bf16
  deform_sample<<<Mrows, 256, 0, stream>>>(val_pad, offb, attnb, refp, samp_bf);

  // 6) final projection -> d_out (fp32)
  {
    const int nrb = Mrows / 64;
    gemm_bf16_wmma_staged<<<nrb * (256 / 64), 256, 0, stream>>>(samp_bf, Wout_t, b_out, out, 256, 0);
  }
}
